// QuantizationLoss_9844065042759
// MI455X (gfx1250) — compile-verified, matrix operand order checked by reading, not run on hardware
//
#include <hip/hip_runtime.h>

typedef __attribute__((ext_vector_type(16))) _Float16 v16h;
typedef __attribute__((ext_vector_type(8)))  _Float16 v8h;
typedef __attribute__((ext_vector_type(8)))  float    v8f;
typedef __attribute__((ext_vector_type(4)))  float    v4f;

#define BLOCK 256          // 8 waves (wave32)
#define MT 4               // point-tiles (16 points each) per wave batch
#define D 64               // feature dim (fixed by reference)
#define LDS_STRIDE 72      // halves per codeword row: 64 + 8 pad (bank-conflict-free b128)

// ---------------------------------------------------------------------------
// Main kernel: fused  dist^2 = xsq + min_j (wsq_j - 2 * <x, w_j>)  via f16 WMMA
// ---------------------------------------------------------------------------
__global__ __launch_bounds__(BLOCK, 1)
void som_qerr_kernel(const float* __restrict__ x,
                     const float* __restrict__ w,
                     float* __restrict__ partials,
                     int N, int HW) {
  extern __shared__ char smem[];
  _Float16* sW  = (_Float16*)smem;                                        // [HW][LDS_STRIDE] f16
  float*    sWsq = (float*)(smem + (size_t)HW * LDS_STRIDE * sizeof(_Float16)); // [HW] f32

  const int tid = threadIdx.x;

  // Stage codebook (f32 -> f16) into padded LDS rows.
  for (int i = tid; i < HW * D; i += BLOCK) {
    const int r = i >> 6;
    const int k = i & 63;
    sW[r * LDS_STRIDE + k] = (_Float16)w[i];
  }
  // Exact f32 squared norms of codewords.
  for (int r = tid; r < HW; r += BLOCK) {
    const float* wr = w + (size_t)r * D;
    float s = 0.f;
#pragma unroll
    for (int k = 0; k < D; ++k) s += wr[k] * wr[k];
    sWsq[r] = s;
  }
  __syncthreads();

  const int lane   = tid & 31;
  const int wv     = tid >> 5;
  const int gwave  = blockIdx.x * (BLOCK >> 5) + wv;
  const int nWaves = gridDim.x * (BLOCK >> 5);
  const int half   = lane >> 4;     // which K-subset this lane holds
  const int lrow   = lane & 15;     // row (A) / column (B) index within tile

  const int tilesTotal = N >> 4;                      // 16 points per tile
  const int batches    = (tilesTotal + MT - 1) / MT;
  const int jTiles     = HW >> 4;                     // codeword tiles of 16

  float waveSum = 0.f;

  for (int b = gwave; b < batches; b += nWaves) {
    const int tile0 = b * MT;

    // Prefetch next batch's x data (emits global_prefetch_b8).
    if (b + nWaves < batches) {
      __builtin_prefetch(x + (size_t)(b + nWaves) * MT * 16 * D, 0, 0);
    }

    // ---- Load A fragments (x tiles, f32 -> f16) + exact f32 xsq ----
    v16h  a[MT][2];
    float xsq[MT];
#pragma unroll
    for (int t = 0; t < MT; ++t) {
      int tile = tile0 + t;
      if (tile >= tilesTotal) tile = tilesTotal - 1;  // clamp; excluded from sum later
      const float* xrow = x + ((size_t)tile * 16 + lrow) * D;
      float ss = 0.f;
#pragma unroll
      for (int f = 0; f < 2; ++f) {          // K 0..31 / 32..63 fragment
#pragma unroll
        for (int c2 = 0; c2 < 2; ++c2) {     // low / high K-chunk within fragment
          const int k0 = f * 32 + c2 * 16 + half * 8;
          const v4f q0 = *(const v4f*)(xrow + k0);
          const v4f q1 = *(const v4f*)(xrow + k0 + 4);
#pragma unroll
          for (int i = 0; i < 4; ++i) {
            a[t][f][c2 * 8 + i]     = (_Float16)q0[i];
            a[t][f][c2 * 8 + 4 + i] = (_Float16)q1[i];
            ss += q0[i] * q0[i] + q1[i] * q1[i];
          }
        }
      }
      ss += __shfl_xor(ss, 16, 32);          // combine half-wave partials
      xsq[t] = ss;                           // lane holds xsq of row (lane & 15)
    }

    // ---- Running min of (wsq - 2*dot) over all codewords ----
    float minv[MT][8];
#pragma unroll
    for (int t = 0; t < MT; ++t)
#pragma unroll
      for (int r = 0; r < 8; ++r) minv[t][r] = 3.4e38f;

    for (int j = 0; j < jTiles; ++j) {
      const int nrow = j * 16 + lrow;
      const _Float16* wrow = sW + (size_t)nrow * LDS_STRIDE;
      const float wsqn = sWsq[nrow];

      v16h bf[2];
#pragma unroll
      for (int f = 0; f < 2; ++f) {
#pragma unroll
        for (int c2 = 0; c2 < 2; ++c2) {
          const int k0 = f * 32 + c2 * 16 + half * 8;
          const v8h h = *(const v8h*)(wrow + k0);   // ds_load_b128
#pragma unroll
          for (int i = 0; i < 8; ++i) bf[f][c2 * 8 + i] = h[i];
        }
      }

#pragma unroll
      for (int t = 0; t < MT; ++t) {
        v8f c = {};
        c = __builtin_amdgcn_wmma_f32_16x16x32_f16(false, a[t][0], false, bf[0],
                                                   (short)0, c, false, false);
        c = __builtin_amdgcn_wmma_f32_16x16x32_f16(false, a[t][1], false, bf[1],
                                                   (short)0, c, false, false);
#pragma unroll
        for (int r = 0; r < 8; ++r)
          minv[t][r] = fminf(minv[t][r], fmaf(c[r], -2.0f, wsqn));
      }
    }

    // ---- Reduce min across the 16 lanes of each half-wave; accumulate sqrt ----
#pragma unroll
    for (int t = 0; t < MT; ++t) {
      const int tile = tile0 + t;
#pragma unroll
      for (int r = 0; r < 8; ++r) {
        float m = minv[t][r];
        m = fminf(m, __shfl_xor(m, 1, 32));
        m = fminf(m, __shfl_xor(m, 2, 32));
        m = fminf(m, __shfl_xor(m, 4, 32));
        m = fminf(m, __shfl_xor(m, 8, 32));
        // row handled by this (half, r): m_row = r + 8*half; fetch its xsq
        const float xr = __shfl(xsq[t], r + half * 8, 32);
        if (lrow == 0 && tile < tilesTotal) {
          waveSum += sqrtf(fmaxf(xr + m, 0.f));
        }
      }
    }
  }

  waveSum += __shfl_xor(waveSum, 16, 32);
  if (lane == 0) partials[gwave] = waveSum;
}

// ---------------------------------------------------------------------------
// Finalize: deterministic reduction of per-wave partials, write mean.
// ---------------------------------------------------------------------------
__global__ void som_finalize_kernel(const float* __restrict__ partials, int count,
                                    float invN, float* __restrict__ out) {
  __shared__ float sh[256];
  float s = 0.f;
  for (int i = threadIdx.x; i < count; i += 256) s += partials[i];
  sh[threadIdx.x] = s;
  __syncthreads();
  for (int off = 128; off > 0; off >>= 1) {
    if ((int)threadIdx.x < off) sh[threadIdx.x] += sh[threadIdx.x + off];
    __syncthreads();
  }
  if (threadIdx.x == 0) out[0] = sh[0] * invN;
}

// ---------------------------------------------------------------------------
extern "C" void kernel_launch(void* const* d_in, const int* in_sizes, int n_in,
                              void* d_out, int out_size, void* d_ws, size_t ws_size,
                              hipStream_t stream) {
  const float* x = (const float*)d_in[0];   // [N, 64] f32
  const float* w = (const float*)d_in[1];   // [32, 32, 64] f32
  const int N  = in_sizes[0] / D;           // 131072
  const int HW = in_sizes[1] / D;           // 1024

  int blocks = 256;
  while (blocks > 1 && (size_t)blocks * (BLOCK / 32) * sizeof(float) > ws_size)
    blocks >>= 1;
  const int nWaves = blocks * (BLOCK / 32);

  const size_t shmem = (size_t)HW * LDS_STRIDE * sizeof(_Float16)
                     + (size_t)HW * sizeof(float);

  som_qerr_kernel<<<blocks, BLOCK, shmem, stream>>>(x, w, (float*)d_ws, N, HW);
  som_finalize_kernel<<<1, 256, 0, stream>>>((const float*)d_ws, nWaves,
                                             1.0f / (float)N, (float*)d_out);
}
